// SparseTransformerLayer_50422916055413
// MI455X (gfx1250) — matrix operand order repaired and checked by reading, compiled, not run on hardware
//
#include <hip/hip_runtime.h>

typedef __attribute__((ext_vector_type(16))) __bf16 v16bf;
typedef __attribute__((ext_vector_type(8)))  __bf16 v8bf;
typedef __attribute__((ext_vector_type(8)))  float  v8f;
typedef __attribute__((ext_vector_type(4)))  unsigned int v4u;
typedef __attribute__((ext_vector_type(8)))  int v8i;
typedef __attribute__((ext_vector_type(4)))  int v4i;

#define BB 4
#define SS 1024
#define DD 1024
#define HH 16
#define HD 64
#define FF 4096
#define EE 8
#define KK 2
#define NT (BB*SS)        /* 4096 tokens */
#define CAPN (NT/2)       /* 2048 */
#define NEGF (-1e30f)
#define KP 40             /* padded LDS K stride (bf16): 80B rows = TDM pad 4dw per 16dw */

#if defined(__has_builtin)
#if __has_builtin(__builtin_amdgcn_tensor_load_to_lds) && __has_builtin(__builtin_amdgcn_s_wait_tensorcnt)
#define USE_TDM 1
#endif
#endif
#ifndef USE_TDM
#define USE_TDM 0
#endif

static __device__ __forceinline__ __bf16 f2bf(float f) {
  unsigned u = __builtin_bit_cast(unsigned, f);
  u += 0x7fffu + ((u >> 16) & 1u);                 // round-to-nearest-even
  return __builtin_bit_cast(__bf16, (unsigned short)(u >> 16));
}

struct GemmArgs {
  const __bf16* A; const __bf16* W; const float* bias; const float* aux;
  const int* tok; const float* wgt;
  float* C; __bf16* Cb; float* scat;
  long long sA1, sA2, sW1, sW2, sB1, sB2, sC1, sC2, sX1, sX2, sT1;
  int lda, ldw, ldc, ldscat, M, Nn, K, zdiv, mode;
  float scale;
};
// mode: 0 = acc+bias ; 1 = acc*scale ; 2 = (acc+bias)*silu(aux) ; 3 = atomic scatter (acc+bias)*wgt[m]

#if USE_TDM
// Issue one TDM 2D tile DMA: 128 rows x 32 bf16 cols, row-major source (ld elems),
// zero-fill beyond (rows_rem, cols_rem), LDS-padded 16B per 64B row -> KP=40 layout.
static __device__ __forceinline__ void tdm_tile(const __bf16* base, long long row0, long long col0,
                                                int ld, long long rows_total, long long cols_total,
                                                unsigned lds_off) {
  unsigned long long ga = (unsigned long long)base + 2ull * ((unsigned long long)row0 * ld + (unsigned long long)col0);
  unsigned td0 = (unsigned)(cols_total - col0);   // tensor dim0 (elems) remaining
  unsigned td1 = (unsigned)(rows_total - row0);   // tensor dim1 (rows) remaining -> OOB rows read 0
  v4u g0;
  g0[0] = 1u;                                     // count=1, user descriptor
  g0[1] = lds_off;                                // lds_addr (bytes)
  g0[2] = (unsigned)(ga & 0xffffffffu);
  g0[3] = (unsigned)((ga >> 32) & 0x01ffffffu) | 0x80000000u;  // type=2 ("image")
  v8i g1;
  g1[0] = (int)0x06D10000;                        // data_size=2B | pad_enable | pad_interval=16dw | pad_amount=4dw
  g1[1] = (int)((td0 & 0xffffu) << 16);           // tensor_dim0[15:0]
  g1[2] = (int)((td0 >> 16) | ((td1 & 0xffffu) << 16));
  g1[3] = (int)((td1 >> 16) | (32u << 16));       // tile_dim0 = 32 elems
  g1[4] = 128;                                    // tile_dim1 = 128 rows, tile_dim2 = 0
  g1[5] = ld;                                     // tensor_dim0_stride (elems)
  g1[6] = 0;
  g1[7] = 0;
  v4i gz4 = {0, 0, 0, 0};
  v8i gz8 = {0, 0, 0, 0, 0, 0, 0, 0};
  __builtin_amdgcn_tensor_load_to_lds(g0, g1, gz4, gz4, gz8, 0);
}
#endif

__global__ __launch_bounds__(256) void gemm_wmma_bf16(GemmArgs g) {
#if USE_TDM
  __shared__ __align__(16) __bf16 smem[2 * 2 * 128 * KP];   // [buf][A|W][128*KP]
#else
  __shared__ __align__(16) __bf16 smem[2 * 128 * KP];       // [A|W][128*KP]
#endif

  const int z = blockIdx.z;
  const long long zq = z / g.zdiv, zr = z % g.zdiv;
  const __bf16* A   = g.A + zq * g.sA1 + zr * g.sA2;
  const __bf16* W   = g.W + zq * g.sW1 + zr * g.sW2;
  const float* bias = g.bias ? g.bias + zq * g.sB1 + zr * g.sB2 : nullptr;
  const float* aux  = g.aux  ? g.aux  + zq * g.sX1 + zr * g.sX2 : nullptr;
  float*  C         = g.C    ? g.C    + zq * g.sC1 + zr * g.sC2 : nullptr;
  __bf16* Cb        = g.Cb   ? g.Cb   + zq * g.sC1 + zr * g.sC2 : nullptr;

  const int m0 = blockIdx.y * 128, n0 = blockIdx.x * 128;
  const int tid  = threadIdx.x;
  const int lane = tid & 31, wv = tid >> 5;
  const int wm = wv >> 2, wn = wv & 3;            // 2x4 wave grid: 64 rows x 32 cols per wave
  const int half = lane >> 4, l15 = lane & 15;
  const int nk = g.K / 32;

  v8f acc[4][2];
#pragma unroll
  for (int mi = 0; mi < 4; ++mi)
#pragma unroll
    for (int ni = 0; ni < 2; ++ni)
#pragma unroll
      for (int r = 0; r < 8; ++r) acc[mi][ni][r] = 0.f;

#if USE_TDM
  const unsigned smem_base = (unsigned)(unsigned long long)&smem[0];
  // prologue: wave0 DMAs tile 0 into buffer 0
  if (wv == 0) {
    tdm_tile(A, m0, 0, g.lda, g.M, g.K, smem_base + 0u);
    tdm_tile(W, n0, 0, g.ldw, g.Nn, g.K, smem_base + 10240u);
    __builtin_amdgcn_s_wait_tensorcnt(0);
  }
  __syncthreads();
#endif

  for (int kt = 0; kt < nk; ++kt) {
#if USE_TDM
    const int cur = kt & 1;
    const __bf16* As = &smem[cur * 2 * 128 * KP];
    const __bf16* Ws = As + 128 * KP;
    if (kt + 1 < nk && wv == 0) {                 // overlap next tile DMA with current WMMAs
      unsigned nb = smem_base + (unsigned)((cur ^ 1) * 20480);
      tdm_tile(A, m0, (long long)(kt + 1) * 32, g.lda, g.M, g.K, nb + 0u);
      tdm_tile(W, n0, (long long)(kt + 1) * 32, g.ldw, g.Nn, g.K, nb + 10240u);
    }
#else
    const int k0 = kt * 32;
    __bf16* As = &smem[0];
    __bf16* Ws = As + 128 * KP;
#pragma unroll
    for (int it = 0; it < 2; ++it) {              // A: 512 chunks of 8 bf16
      int li = it * 256 + tid;
      int r = li >> 2, kq = (li & 3) * 8;
      *(v8bf*)&As[r * KP + kq] = *(const v8bf*)(A + (long long)(m0 + r) * g.lda + k0 + kq);
    }
#pragma unroll
    for (int it = 0; it < 2; ++it) {              // W: zero-fill OOB rows
      int li = it * 256 + tid;
      int r = li >> 2, kq = (li & 3) * 8;
      v8bf vz = {};
      if (n0 + r < g.Nn) vz = *(const v8bf*)(W + (long long)(n0 + r) * g.ldw + k0 + kq);
      *(v8bf*)&Ws[r * KP + kq] = vz;
    }
    __syncthreads();
#endif

    v16bf afr[4], bfr[2];
#pragma unroll
    for (int mi = 0; mi < 4; ++mi) {
      int row = wm * 64 + mi * 16 + l15;          // A frag: K = (e&7) + 8*half + 16*(e>>3)
      v8bf lo = *(const v8bf*)&As[row * KP + 8 * half];
      v8bf hi = *(const v8bf*)&As[row * KP + 16 + 8 * half];
      afr[mi] = __builtin_shufflevector(lo, hi, 0,1,2,3,4,5,6,7,8,9,10,11,12,13,14,15);
    }
#pragma unroll
    for (int ni = 0; ni < 2; ++ni) {
      int row = wn * 32 + ni * 16 + l15;          // B frag: K = (e&7) + 8*half + 16*(e>>3)
      v8bf lo = *(const v8bf*)&Ws[row * KP + 8 * half];
      v8bf hi = *(const v8bf*)&Ws[row * KP + 16 + 8 * half];
      bfr[ni] = __builtin_shufflevector(lo, hi, 0,1,2,3,4,5,6,7,8,9,10,11,12,13,14,15);
    }
#pragma unroll
    for (int mi = 0; mi < 4; ++mi)
#pragma unroll
      for (int ni = 0; ni < 2; ++ni)
        acc[mi][ni] = __builtin_amdgcn_wmma_f32_16x16x32_bf16(
            false, afr[mi], false, bfr[ni], (short)0, acc[mi][ni], false, false);

#if USE_TDM
    if (wv == 0 && kt + 1 < nk) __builtin_amdgcn_s_wait_tensorcnt(0);
    __syncthreads();
#else
    __syncthreads();
#endif
  }

  // epilogue
  const int* tok  = (g.mode == 3) ? g.tok + (long long)z * g.sT1 : nullptr;
  const float* wg = (g.mode == 3) ? g.wgt + (long long)z * g.sT1 : nullptr;
#pragma unroll
  for (int ni = 0; ni < 2; ++ni) {
    int n = n0 + wn * 32 + ni * 16 + l15;
    if (n >= g.Nn) continue;
    float bv = bias ? bias[n] : 0.f;
#pragma unroll
    for (int mi = 0; mi < 4; ++mi) {
#pragma unroll
      for (int r = 0; r < 8; ++r) {
        int m = m0 + wm * 64 + mi * 16 + r + 8 * half;
        float v = acc[mi][ni][r];
        if (g.mode == 3) {
          float wt = wg[m];
          if (wt != 0.f)
            __hip_atomic_fetch_add(&g.scat[(long long)tok[m] * g.ldscat + n],
                                   (v + bv) * wt, __ATOMIC_RELAXED, __HIP_MEMORY_SCOPE_AGENT);
          continue;
        }
        float outv;
        if (g.mode == 0) outv = v + bv;
        else if (g.mode == 1) outv = v * g.scale;
        else { float gt = aux[(long long)m * g.ldc + n];
               outv = (v + bv) * (gt / (1.f + __expf(-gt))); }
        if (C)  C [(long long)m * g.ldc + n] = outv;
        if (Cb) Cb[(long long)m * g.ldc + n] = f2bf(outv);
      }
    }
  }
}

__global__ void convert_bf16(const float* s, __bf16* d, long long n4) {
  long long i = (long long)blockIdx.x * blockDim.x + threadIdx.x;
  if (i >= n4) return;
  float4 f = ((const float4*)s)[i];
  __bf16* p = d + i * 4;
  p[0] = f2bf(f.x); p[1] = f2bf(f.y); p[2] = f2bf(f.z); p[3] = f2bf(f.w);
}

__global__ void rope_kernel(const float* qkv, __bf16* qkvb) {
  int idx = blockIdx.x * blockDim.x + threadIdx.x;   // over NT*HH*32
  if (idx >= NT * HH * 32) return;
  int j = idx & 31;
  int h = (idx >> 5) & (HH - 1);
  int t = idx >> 9;
  int s = t & (SS - 1);
  float inv = __expf(-0.28782313662425572f * (float)j);  // ln(10000)/32
  float ang = (float)s * inv;
  float c = cosf(ang), sn = sinf(ang);
  long long base = (long long)t * (3 * DD) + h * HD + j;
  float q1 = qkv[base], q2 = qkv[base + 32];
  qkvb[base]      = f2bf(q1 * c - q2 * sn);
  qkvb[base + 32] = f2bf(q2 * c + q1 * sn);
  float k1 = qkv[base + DD], k2 = qkv[base + DD + 32];
  qkvb[base + DD]      = f2bf(k1 * c - k2 * sn);
  qkvb[base + DD + 32] = f2bf(k2 * c + k1 * sn);
}

__global__ __launch_bounds__(256) void softmax_rows(float* x, __bf16* xb, int ncols) {
  __shared__ float red[256];
  long long row = blockIdx.x;
  float* p = x + row * ncols;
  __bf16* pb = xb + row * ncols;
  int tid = threadIdx.x;
  float mx = -3.4e38f;
  for (int j = tid; j < ncols; j += 256) mx = fmaxf(mx, p[j]);
  red[tid] = mx; __syncthreads();
  for (int s = 128; s > 0; s >>= 1) { if (tid < s) red[tid] = fmaxf(red[tid], red[tid + s]); __syncthreads(); }
  mx = red[0]; __syncthreads();
  float sum = 0.f;
  for (int j = tid; j < ncols; j += 256) { float e = __expf(p[j] - mx); p[j] = e; sum += e; }
  red[tid] = sum; __syncthreads();
  for (int s = 128; s > 0; s >>= 1) { if (tid < s) red[tid] += red[tid + s]; __syncthreads(); }
  float inv = 1.f / red[0];
  for (int j = tid; j < ncols; j += 256) { float v = p[j] * inv; p[j] = v; pb[j] = f2bf(v); }
}

__global__ void transpose_v(const float* vh, __bf16* vhT) {
  int idx = blockIdx.x * blockDim.x + threadIdx.x;   // over NT*DD
  if (idx >= NT * DD) return;
  int d = idx & (HD - 1);
  int h = (idx >> 6) & (HH - 1);
  int t = idx >> 10;
  int b = t >> 10, s = t & (SS - 1);
  vhT[(((long long)(b * HH + h) * HD + d) * SS) + s] = f2bf(vh[idx]);
}

__global__ __launch_bounds__(256) void add_ln(const float* x, const float* r,
                                              const float* gm, const float* bt, float* out) {
  __shared__ float red[256];
  long long row = blockIdx.x;
  const float* xp = x + row * DD; const float* rp = r + row * DD;
  float* op = out + row * DD;
  int tid = threadIdx.x;
  float vals[4]; float ls = 0.f;
#pragma unroll
  for (int i = 0; i < 4; ++i) { int j = tid + i * 256; vals[i] = xp[j] + rp[j]; ls += vals[i]; }
  red[tid] = ls; __syncthreads();
  for (int s = 128; s > 0; s >>= 1) { if (tid < s) red[tid] += red[tid + s]; __syncthreads(); }
  float mean = red[0] * (1.f / DD); __syncthreads();
  float lv = 0.f;
#pragma unroll
  for (int i = 0; i < 4; ++i) { float d = vals[i] - mean; lv += d * d; }
  red[tid] = lv; __syncthreads();
  for (int s = 128; s > 0; s >>= 1) { if (tid < s) red[tid] += red[tid + s]; __syncthreads(); }
  float rstd = rsqrtf(red[0] * (1.f / DD) + 1e-5f);
#pragma unroll
  for (int i = 0; i < 4; ++i) { int j = tid + i * 256; op[j] = (vals[i] - mean) * rstd * gm[j] + bt[j]; }
}

__global__ __launch_bounds__(256) void router_kernel(const float* x1, const float* rw,
                                                     const float* rb, float* logits) {
  __shared__ float xrow[DD];
  __shared__ float red[256];
  int t = blockIdx.x, tid = threadIdx.x;
#pragma unroll
  for (int i = 0; i < 4; ++i) xrow[tid + i * 256] = x1[(long long)t * DD + tid + i * 256];
  __syncthreads();
  for (int e = 0; e < EE; ++e) {
    float s = 0.f;
    for (int j = tid; j < DD; j += 256) s += xrow[j] * rw[e * DD + j];
    red[tid] = s; __syncthreads();
    for (int st = 128; st > 0; st >>= 1) { if (tid < st) red[tid] += red[tid + st]; __syncthreads(); }
    if (tid == 0) logits[(long long)t * EE + e] = red[0] + rb[e];
    __syncthreads();
  }
}

__global__ void top2_kernel(const float* logits, float* topv, int* topi) {
  int t = blockIdx.x * blockDim.x + threadIdx.x;
  if (t >= NT) return;
  float l[EE]; float mx = -3.4e38f;
  for (int e = 0; e < EE; ++e) { l[e] = logits[(long long)t * EE + e]; mx = fmaxf(mx, l[e]); }
  float sum = 0.f;
  for (int e = 0; e < EE; ++e) { l[e] = __expf(l[e] - mx); sum += l[e]; }
  float inv = 1.f / sum;
  for (int e = 0; e < EE; ++e) l[e] *= inv;
  int i1 = 0;
  for (int e = 1; e < EE; ++e) if (l[e] > l[i1]) i1 = e;
  int i2 = (i1 == 0) ? 1 : 0;
  for (int e = 0; e < EE; ++e) if (e != i1 && l[e] > l[i2]) i2 = e;
  topi[t * 2] = i1; topi[t * 2 + 1] = i2;
  topv[t * 2] = l[i1]; topv[t * 2 + 1] = l[i2];
}

__global__ __launch_bounds__(256) void capacity_kernel(const float* logits, const int* topi,
                                                       const float* topv, int* tok, float* wgt) {
  __shared__ float sc[NT];
  int e = blockIdx.x, tid = threadIdx.x;
  for (int c = tid; c < CAPN; c += 256) { tok[e * CAPN + c] = 0; wgt[e * CAPN + c] = 0.f; }
  for (int i = tid; i < NT; i += 256) {
    bool cand = (topi[2 * i] == e) || (topi[2 * i + 1] == e);
    sc[i] = cand ? logits[(long long)i * EE + e] : NEGF;
  }
  __syncthreads();
  for (int i = tid; i < NT; i += 256) {
    float my = sc[i];
    if (my > NEGF * 0.5f) {
      int rank = 0;
      for (int j = 0; j < NT; ++j) rank += (sc[j] > my) || (sc[j] == my && j < i);
      if (rank < CAPN) {
        tok[e * CAPN + rank] = i;
        int pos = (topi[2 * i] == e) ? 0 : 1;
        wgt[e * CAPN + rank] = topv[i * 2 + pos];
      }
    }
  }
}

__global__ void gather_xt(const float* x1, const int* tok, __bf16* xt) {
  long long idx = (long long)blockIdx.x * blockDim.x + threadIdx.x;  // over E*CAP*D/4
  if (idx >= (long long)EE * CAPN * DD / 4) return;
  int d4 = (int)(idx % (DD / 4));
  long long ec = idx / (DD / 4);
  int c = (int)(ec % CAPN), e = (int)(ec / CAPN);
  int t = tok[e * CAPN + c];
  float4 f = ((const float4*)(x1 + (long long)t * DD))[d4];
  __bf16* p = xt + idx * 4;
  p[0] = f2bf(f.x); p[1] = f2bf(f.y); p[2] = f2bf(f.z); p[3] = f2bf(f.w);
}

__global__ void zero_f32(float* p, long long n) {
  long long i = (long long)blockIdx.x * blockDim.x + threadIdx.x;
  if (i < n) p[i] = 0.f;
}

extern "C" void kernel_launch(void* const* d_in, const int* in_sizes, int n_in,
                              void* d_out, int out_size, void* d_ws, size_t ws_size,
                              hipStream_t stream) {
  const float* x          = (const float*)d_in[0];
  const float* in_proj_w  = (const float*)d_in[1];
  const float* in_proj_b  = (const float*)d_in[2];
  const float* out_proj_w = (const float*)d_in[3];
  const float* out_proj_b = (const float*)d_in[4];
  const float* norm1_g    = (const float*)d_in[5];
  const float* norm1_b    = (const float*)d_in[6];
  const float* norm2_g    = (const float*)d_in[7];
  const float* norm2_b    = (const float*)d_in[8];
  const float* router_w   = (const float*)d_in[9];
  const float* router_b   = (const float*)d_in[10];
  const float* gate_w     = (const float*)d_in[11];
  const float* gate_b     = (const float*)d_in[12];
  const float* val_w      = (const float*)d_in[13];
  const float* val_b      = (const float*)d_in[14];
  const float* wo_w       = (const float*)d_in[15];
  const float* wo_b       = (const float*)d_in[16];
  (void)in_sizes; (void)n_in; (void)out_size; (void)ws_size;

  float* y      = (float*)d_out;                       // [N,D] f32
  float* logits = y + (size_t)NT * DD;                 // [N,E] f32
  int*   topi   = (int*)(logits + (size_t)NT * EE);    // [N,2] i32

  char* wsb = (char*)d_ws; size_t off = 0;
  auto allocF = [&](size_t elems) -> float* {
    float* p = (float*)(wsb + off); off += ((elems * 4) + 255) & ~(size_t)255; return p; };
  auto allocH = [&](size_t elems) -> __bf16* {
    __bf16* p = (__bf16*)(wsb + off); off += ((elems * 2) + 255) & ~(size_t)255; return p; };

  float*  qkv    = allocF((size_t)NT * 3 * DD);
  float*  vh     = allocF((size_t)NT * DD);
  float*  scores = allocF((size_t)BB * HH * SS * SS);
  float*  attn_o = allocF((size_t)NT * DD);
  float*  x1     = allocF((size_t)NT * DD);
  float*  G      = allocF((size_t)EE * CAPN * FF);
  float*  ffn    = allocF((size_t)NT * DD);
  float*  topv   = allocF((size_t)NT * KK);
  float*  wbuf   = allocF((size_t)EE * CAPN);
  int*    tokbuf = (int*)allocF((size_t)EE * CAPN);

  __bf16* x_b    = allocH((size_t)NT * DD);
  __bf16* qkv_b  = allocH((size_t)NT * 3 * DD);
  __bf16* qh_b   = allocH((size_t)NT * DD);
  __bf16* kh_b   = allocH((size_t)NT * DD);
  __bf16* vhT_b  = allocH((size_t)NT * DD);
  __bf16* prob_b = allocH((size_t)BB * HH * SS * SS);
  __bf16* o_b    = allocH((size_t)NT * DD);
  __bf16* xt_b   = allocH((size_t)EE * CAPN * DD);
  __bf16* H_b    = allocH((size_t)EE * CAPN * FF);
  __bf16* ipw_b  = allocH((size_t)3 * DD * DD);
  __bf16* opw_b  = allocH((size_t)DD * DD);
  __bf16* gw_b   = allocH((size_t)EE * FF * DD);
  __bf16* vw_b   = allocH((size_t)EE * FF * DD);
  __bf16* wow_b  = allocH((size_t)EE * DD * FF);

  auto conv = [&](const float* s, __bf16* d, long long n) {
    convert_bf16<<<(unsigned)((n / 4 + 255) / 256), 256, 0, stream>>>(s, d, n / 4); };
  conv(x, x_b, (long long)NT * DD);
  conv(in_proj_w, ipw_b, (long long)3 * DD * DD);
  conv(out_proj_w, opw_b, (long long)DD * DD);
  conv(gate_w, gw_b, (long long)EE * FF * DD);
  conv(val_w, vw_b, (long long)EE * FF * DD);
  conv(wo_w, wow_b, (long long)EE * DD * FF);

  auto run = [&](const GemmArgs& g, int Z) {
    dim3 grid((g.Nn + 127) / 128, (g.M + 127) / 128, Z);
    gemm_wmma_bf16<<<grid, 256, 0, stream>>>(g);
  };

  // 1) qkv = x @ in_proj_w^T + b   (f32 for RoPE + bf16 mirror for v-path)
  { GemmArgs g{}; g.A = x_b; g.lda = DD; g.W = ipw_b; g.ldw = DD; g.bias = in_proj_b;
    g.C = qkv; g.Cb = qkv_b; g.ldc = 3 * DD; g.M = NT; g.Nn = 3 * DD; g.K = DD; g.zdiv = 1; g.mode = 0; run(g, 1); }

  // 2) RoPE: read f32 q/k, write bf16 q/k into qkv_b (v part already mirrored)
  rope_kernel<<<(NT * HH * 32) / 256, 256, 0, stream>>>(qkv, qkv_b);

  // 3) qh/kh/vh second projections
  { GemmArgs g{}; g.A = qkv_b; g.lda = 3 * DD; g.W = ipw_b; g.ldw = DD; g.bias = in_proj_b;
    g.Cb = qh_b; g.ldc = DD; g.M = NT; g.Nn = DD; g.K = DD; g.zdiv = 1; g.mode = 0; run(g, 1); }
  { GemmArgs g{}; g.A = qkv_b + DD; g.lda = 3 * DD; g.W = ipw_b + (size_t)DD * DD; g.ldw = DD;
    g.bias = in_proj_b + DD; g.Cb = kh_b; g.ldc = DD; g.M = NT; g.Nn = DD; g.K = DD; g.zdiv = 1; g.mode = 0; run(g, 1); }
  { GemmArgs g{}; g.A = qkv_b + 2 * DD; g.lda = 3 * DD; g.W = ipw_b + (size_t)2 * DD * DD; g.ldw = DD;
    g.bias = in_proj_b + 2 * DD; g.C = vh; g.ldc = DD; g.M = NT; g.Nn = DD; g.K = DD; g.zdiv = 1; g.mode = 0; run(g, 1); }

  // 4) scores[b,h] = (qh @ kh^T) / sqrt(HD)   Z = B*H
  { GemmArgs g{}; g.A = qh_b; g.sA1 = (long long)SS * DD; g.sA2 = HD; g.lda = DD;
    g.W = kh_b; g.sW1 = (long long)SS * DD; g.sW2 = HD; g.ldw = DD;
    g.C = scores; g.sC1 = (long long)HH * SS * SS; g.sC2 = (long long)SS * SS; g.ldc = SS;
    g.M = SS; g.Nn = SS; g.K = HD; g.zdiv = HH; g.mode = 1; g.scale = 0.125f; run(g, BB * HH); }

  // 5) softmax rows (f32 in place + bf16 probs for PV GEMM)
  softmax_rows<<<BB * HH * SS, 256, 0, stream>>>(scores, prob_b, SS);

  // 6) vhT[b,h,d,s] = vh[b,s,h,d] (bf16)
  transpose_v<<<(NT * DD) / 256, 256, 0, stream>>>(vh, vhT_b);

  // 7) o = attn @ vh   Z = B*H
  { GemmArgs g{}; g.A = prob_b; g.sA1 = (long long)HH * SS * SS; g.sA2 = (long long)SS * SS; g.lda = SS;
    g.W = vhT_b; g.sW1 = (long long)HH * HD * SS; g.sW2 = (long long)HD * SS; g.ldw = SS;
    g.Cb = o_b; g.sC1 = (long long)SS * DD; g.sC2 = HD; g.ldc = DD;
    g.M = SS; g.Nn = HD; g.K = SS; g.zdiv = HH; g.mode = 0; run(g, BB * HH); }

  // 8) out_proj
  { GemmArgs g{}; g.A = o_b; g.lda = DD; g.W = opw_b; g.ldw = DD; g.bias = out_proj_b;
    g.C = attn_o; g.ldc = DD; g.M = NT; g.Nn = DD; g.K = DD; g.zdiv = 1; g.mode = 0; run(g, 1); }

  // 9) x1 = LN(x + attn_o)
  add_ln<<<NT, 256, 0, stream>>>(x, attn_o, norm1_g, norm1_b, x1);

  // 10-11) router logits + top-2 (straight into d_out slices)
  router_kernel<<<NT, 256, 0, stream>>>(x1, router_w, router_b, logits);
  top2_kernel<<<NT / 256, 256, 0, stream>>>(logits, topv, topi);

  // 12-14) capacity-limited dispatch
  zero_f32<<<(NT * DD) / 256, 256, 0, stream>>>(ffn, (long long)NT * DD);
  capacity_kernel<<<EE, 256, 0, stream>>>(logits, topi, topv, tokbuf, wbuf);
  gather_xt<<<(EE * CAPN * DD / 4) / 256, 256, 0, stream>>>(x1, tokbuf, xt_b);

  // 15) gate GEMM: G = xt @ gate_w^T + gate_b      Z = E
  { GemmArgs g{}; g.A = xt_b; g.sA1 = (long long)CAPN * DD; g.lda = DD;
    g.W = gw_b; g.sW1 = (long long)FF * DD; g.ldw = DD;
    g.bias = gate_b; g.sB1 = FF;
    g.C = G; g.sC1 = (long long)CAPN * FF; g.ldc = FF;
    g.M = CAPN; g.Nn = FF; g.K = DD; g.zdiv = 1; g.mode = 0; run(g, EE); }

  // 16) val GEMM fused SwiGLU -> bf16 H
  { GemmArgs g{}; g.A = xt_b; g.sA1 = (long long)CAPN * DD; g.lda = DD;
    g.W = vw_b; g.sW1 = (long long)FF * DD; g.ldw = DD;
    g.bias = val_b; g.sB1 = FF;
    g.aux = G; g.sX1 = (long long)CAPN * FF;
    g.Cb = H_b; g.sC1 = (long long)CAPN * FF; g.ldc = FF;
    g.M = CAPN; g.Nn = FF; g.K = DD; g.zdiv = 1; g.mode = 2; run(g, EE); }

  // 17) wo GEMM with weighted atomic scatter-add combine
  { GemmArgs g{}; g.A = H_b; g.sA1 = (long long)CAPN * FF; g.lda = FF;
    g.W = wow_b; g.sW1 = (long long)DD * FF; g.ldw = FF;
    g.bias = wo_b; g.sB1 = DD;
    g.tok = tokbuf; g.wgt = wbuf; g.sT1 = CAPN;
    g.scat = ffn; g.ldscat = DD;
    g.M = CAPN; g.Nn = DD; g.K = FF; g.zdiv = 1; g.mode = 3; run(g, EE); }

  // 18) y = LN(x1 + ffn)
  add_ln<<<NT, 256, 0, stream>>>(x1, ffn, norm2_g, norm2_b, y);
}
